// KPConvSimpleBlock_67714454389185
// MI455X (gfx1250) — matrix operand order repaired and checked by reading, not compiled, not run on hardware
//
#include <hip/hip_runtime.h>

// ---------------------------------------------------------------------------
// KPConv simple block for gfx1250 (MI455X): ball query -> KPConv via
// v_wmma_f32_16x16x32_bf16 -> BatchNorm(train stats) + LeakyReLU.
// ---------------------------------------------------------------------------

typedef __bf16  v8bf  __attribute__((ext_vector_type(8)));
typedef __bf16  v16bf __attribute__((ext_vector_type(16)));
typedef float   v8f   __attribute__((ext_vector_type(8)));

namespace {
constexpr int B_   = 8;
constexpr int N_   = 4096;
constexpr int CIN  = 64;
constexpr int COUT = 128;
constexpr int KPN  = 15;     // kernel points
constexpr int NN   = 34;     // max neighbors
constexpr int M    = B_ * N_;            // 32768 points
constexpr float R2      = 0.01f;         // RADIUS^2
constexpr float INV_EXT = 25.0f;         // 1 / KP_EXTENT (0.04)
constexpr int LIST_MAX  = 64;            // in-radius candidate cap (P(>64) ~ 0)

// ---- workspace layout (bytes) ----
constexpr size_t WS_BTILE  = 0;                                   // 8*32*32*16*2 = 262144
constexpr size_t WS_NIDX   = 262144;                              // M*NN*4      = 4456448
constexpr size_t WS_PREOUT = WS_NIDX + (size_t)M * NN * 4;        // M*COUT*4    = 16777216
constexpr size_t WS_PART   = WS_PREOUT + (size_t)M * COUT * 4;    // 2*256*128*4 = 262144
constexpr size_t WS_SS     = WS_PART + 2ull * 256 * 128 * 4;      // scale+shift = 1024

// ---- dynamic LDS layout for the WMMA kernel (per WG of 16 queries) ----
constexpr int PAIRS    = 16 * NN;                     // 544 (query, neighbor) pairs
constexpr int REL_OFF  = 0;                           // 544 * float4
constexpr int NF_OFF   = REL_OFF + PAIRS * 4 * 4;     // 8704
constexpr int WF_ROW_E = 1040;                        // bf16 elems/row (1024 + pad)
constexpr int WF_OFF   = NF_OFF + PAIRS * CIN * 4;    // 147968
constexpr int LDS_BYTES = WF_OFF + 16 * WF_ROW_E * 2; // 181248 (< 320KB WGP LDS)
}

__device__ __forceinline__ unsigned short f2bf(float x) {
  unsigned u = __float_as_uint(x);
  u += 0x7FFFu + ((u >> 16) & 1u);     // round-to-nearest-even
  return (unsigned short)(u >> 16);
}
__device__ __forceinline__ unsigned packbf(float a, float b) {
  return (unsigned)f2bf(a) | ((unsigned)f2bf(b) << 16);
}

// ---------------------------------------------------------------------------
// Kernel 1: repack weight [15][64][128] f32 -> bf16 B-tiles in WMMA-B order.
// Layout: [ntile(8)][kstep(32)][lane(32)][elem(16)] ; lane<16 holds K 0..15,
// lane>=16 holds K 16..31 of the 32-deep step; N = ntile*16 + (lane&15).
// K rows 960..1023 zero-padded.
// ---------------------------------------------------------------------------
__global__ __launch_bounds__(256) void krepack(const float* __restrict__ weight,
                                               unsigned short* __restrict__ btile) {
  int g = blockIdx.x * 256 + threadIdx.x;   // 131072 outputs
  if (g >= 8 * 32 * 32 * 16) return;
  int e    = g & 15;
  int lane = (g >> 4) & 31;
  int ks   = (g >> 9) & 31;
  int nt   = g >> 14;
  int n  = nt * 16 + (lane & 15);
  int kk = ks * 32 + ((lane >> 4) << 4) + e;
  float v = (kk < KPN * CIN) ? weight[kk * COUT + n] : 0.0f;
  btile[g] = f2bf(v);
}

// ---------------------------------------------------------------------------
// Kernel 2: ball query. One wave per query point; cloud xyz staged in LDS.
// Writes neighbor flat indices [M][NN]; shadow = M.
// ---------------------------------------------------------------------------
__global__ __launch_bounds__(256) void kball(const float* __restrict__ xyz,
                                             int* __restrict__ nidx) {
  __shared__ __align__(16) float spts[N_ * 3];
  __shared__ float listd[8][LIST_MAX];
  __shared__ int   listi[8][LIST_MAX];

  const int t     = threadIdx.x;
  const int blk   = blockIdx.x;           // 4096 blocks, 8 queries each
  const int cloud = blk >> 9;             // 512 blocks per cloud
  const float4* src4 = (const float4*)(xyz + (size_t)cloud * N_ * 3);
  float4* dst4 = (float4*)spts;
  for (int i = t; i < N_ * 3 / 4; i += 256) dst4[i] = src4[i];
  __syncthreads();

  const int wid = t >> 5, lane = t & 31;
  const int qlocal = (blk & 511) * 8 + wid;
  const int m = cloud * N_ + qlocal;
  const float qx = spts[qlocal * 3 + 0];
  const float qy = spts[qlocal * 3 + 1];
  const float qz = spts[qlocal * 3 + 2];

  int cnt = 0;
  const unsigned lt = (1u << lane) - 1u;
  for (int c0 = 0; c0 < N_; c0 += 32) {
    int i = c0 + lane;
    float dx = spts[i * 3 + 0] - qx;
    float dy = spts[i * 3 + 1] - qy;
    float dz = spts[i * 3 + 2] - qz;
    float d2 = dx * dx + dy * dy + dz * dz;
    bool in = d2 <= R2;
    unsigned msk = __builtin_amdgcn_ballot_w32(in);
    if (in) {
      int slot = cnt + __popc(msk & lt);
      if (slot < LIST_MAX) { listd[wid][slot] = d2; listi[wid][slot] = cloud * N_ + i; }
    }
    cnt += __popc(msk);
    if (cnt > LIST_MAX) cnt = LIST_MAX;
  }

  if (cnt <= NN) {
    for (int s = lane; s < NN; s += 32)
      nidx[(size_t)m * NN + s] = (s < cnt) ? listi[wid][s] : M;
  } else {
    // extract the 34 smallest d2, deterministically (tie-break on slot).
    for (int it = 0; it < NN; ++it) {
      float bd = 1e30f; int bs = -1;
      for (int s = lane; s < cnt; s += 32) {
        float d = listd[wid][s];
        if (d < bd) { bd = d; bs = s; }
      }
      #pragma unroll
      for (int off = 16; off; off >>= 1) {
        float od = __shfl_xor(bd, off, 32);
        int   os = __shfl_xor(bs, off, 32);
        if (od < bd || (od == bd && os < bs)) { bd = od; bs = os; }
      }
      if (lane == 0) {
        nidx[(size_t)m * NN + it] = listi[wid][bs];
        listd[wid][bs] = 1e30f;
      }
    }
  }
}

// ---------------------------------------------------------------------------
// Kernel 3: fused KPConv + GEMM (WMMA). One WG = 16 query points, 8 waves.
// Phase A: gather neighbor feats/relcoords to LDS; each wave accumulates wf
//          for 2 kernel points in fp32 regs, writes bf16 wf tile to LDS.
// Phase B: each wave computes one 16x16 output N-tile with
//          32 x v_wmma_f32_16x16x32_bf16 over the padded K=1024 dim.
// ---------------------------------------------------------------------------
__global__ __launch_bounds__(256) void kkpconv(const float* __restrict__ xyz,
                                               const float* __restrict__ feats,
                                               const float* __restrict__ kp,
                                               const int* __restrict__ nidx,
                                               const unsigned short* __restrict__ btile,
                                               float* __restrict__ preout) {
  extern __shared__ char smem[];
  float* rel = (float*)(smem + REL_OFF);   // [544][4]  (dx,dy,dz,0)
  float* nf  = (float*)(smem + NF_OFF);    // [544][64] fp32 gathered feats
  char*  wf  = smem + WF_OFF;              // [16][1040] bf16 (K padded to 1024)

  const int t = threadIdx.x, lane = t & 31, wid = t >> 5;
  const int m0 = blockIdx.x * 16;

  // ---- Phase A0: gather ----
  for (int p = t; p < PAIRS; p += 256) {
    int q = p / NN, j = p - q * NN;
    int mq = m0 + q;
    float qx = xyz[mq * 3 + 0], qy = xyz[mq * 3 + 1], qz = xyz[mq * 3 + 2];
    int idx = nidx[(size_t)mq * NN + j];
    float4* drow = (float4*)(nf + p * CIN);
    float rx, ry, rz;
    if (idx < M) {
      rx = xyz[idx * 3 + 0] - qx;
      ry = xyz[idx * 3 + 1] - qy;
      rz = xyz[idx * 3 + 2] - qz;
      const float4* srow = (const float4*)(feats + (size_t)idx * CIN);
      #pragma unroll
      for (int v = 0; v < CIN / 4; ++v) drow[v] = srow[v];
    } else {                               // shadow point: far away, zero feats
      rx = 1e6f - qx; ry = 1e6f - qy; rz = 1e6f - qz;
      float4 z = make_float4(0.f, 0.f, 0.f, 0.f);
      #pragma unroll
      for (int v = 0; v < CIN / 4; ++v) drow[v] = z;
    }
    ((float4*)rel)[p] = make_float4(rx, ry, rz, 0.f);
  }
  __syncthreads();

  // ---- Phase A1: wf accumulation (wave wid owns kernel points k0,k1) ----
  const int k0 = wid * 2, k1 = k0 + 1;
  const bool k1ok = (k1 < KPN);
  const int k1c = k1ok ? k1 : (KPN - 1);
  const int ksel = lane >> 4;              // lane<16 -> k0, lane>=16 -> k1
  const float mkx = kp[(ksel ? k1c : k0) * 3 + 0];
  const float mky = kp[(ksel ? k1c : k0) * 3 + 1];
  const float mkz = kp[(ksel ? k1c : k0) * 3 + 2];
  const int qm = lane & 15;                // each lane computes w for (qm, ksel)

  float a0[16], a1[16], b0[16], b1[16];    // [q] x {k0,k1} x {2lane, 2lane+1}
  #pragma unroll
  for (int q = 0; q < 16; ++q) { a0[q] = 0.f; a1[q] = 0.f; b0[q] = 0.f; b1[q] = 0.f; }

  for (int j = 0; j < NN; ++j) {
    float4 r = ((const float4*)rel)[qm * NN + j];
    float dx = r.x - mkx, dy = r.y - mky, dz = r.z - mkz;
    float d = sqrtf(dx * dx + dy * dy + dz * dz);
    float w = fmaxf(0.f, 1.f - d * INV_EXT);   // linear KP influence
    #pragma unroll
    for (int q = 0; q < 16; ++q) {
      float w0 = __shfl(w, q, 32);
      float w1 = __shfl(w, q + 16, 32);
      float2 f = *(const float2*)(nf + (q * NN + j) * CIN + 2 * lane);
      a0[q] = fmaf(w0, f.x, a0[q]);
      a1[q] = fmaf(w0, f.y, a1[q]);
      b0[q] = fmaf(w1, f.x, b0[q]);
      b1[q] = fmaf(w1, f.y, b1[q]);
    }
  }
  #pragma unroll
  for (int q = 0; q < 16; ++q) {
    char* row = wf + q * (WF_ROW_E * 2);
    *(unsigned*)(row + (k0 * CIN + 2 * lane) * 2) = packbf(a0[q], a1[q]);
    *(unsigned*)(row + (k1 * CIN + 2 * lane) * 2) = k1ok ? packbf(b0[q], b1[q]) : 0u; // zero K-pad
  }
  __syncthreads();

  // ---- Phase B: GEMM out[16,128] = wf[16,1024] x W[1024,128] ----
  v8f acc = {0.f, 0.f, 0.f, 0.f, 0.f, 0.f, 0.f, 0.f};
  const int row  = lane & 15;
  const int koff = (lane >> 4) ? 16 : 0;   // contiguous K half-select (matches B tiles)
  const char* wrow = wf + row * (WF_ROW_E * 2);
  const char* bp = (const char*)btile + (((size_t)wid * 32) * 32 + lane) * 32;
  for (int ks = 0; ks < 32; ++ks) {
    v8bf alo = *(const v8bf*)(wrow + (ks * 32 + koff) * 2);
    v8bf ahi = *(const v8bf*)(wrow + (ks * 32 + koff + 8) * 2);
    v16bf A = __builtin_shufflevector(alo, ahi, 0,1,2,3,4,5,6,7,8,9,10,11,12,13,14,15);
    v8bf blo = *(const v8bf*)bp;
    v8bf bhi = *(const v8bf*)(bp + 16);
    v16bf Bm = __builtin_shufflevector(blo, bhi, 0,1,2,3,4,5,6,7,8,9,10,11,12,13,14,15);
    acc = __builtin_amdgcn_wmma_f32_16x16x32_bf16(false, A, false, Bm,
                                                  (short)0, acc, false, false);
    bp += 1024;
    __builtin_prefetch(bp + 2048, 0, 1);   // global_prefetch next B steps
  }
  // C layout: lane<16 -> col=lane, rows v..; lane>=16 -> col=lane-16, rows v+8
  const int coln  = wid * 16 + (lane & 15);
  const int rbase = m0 + ((lane >> 4) ? 8 : 0);
  #pragma unroll
  for (int v = 0; v < 8; ++v)
    preout[(size_t)(rbase + v) * COUT + coln] = acc[v];
}

// ---------------------------------------------------------------------------
// Kernel 4: deterministic per-channel partial sums / sum-of-squares.
// 256 blocks, block p reduces rows [p*128, p*128+128).
// ---------------------------------------------------------------------------
__global__ __launch_bounds__(256) void kstats_part(const float* __restrict__ preout,
                                                   float* __restrict__ part) {
  __shared__ float sh[512];
  const int t = threadIdx.x;
  const int c = t & 127, h = t >> 7;
  const int rend = (blockIdx.x + 1) * 128;
  float s = 0.f, ss = 0.f;
  for (int r = blockIdx.x * 128 + h; r < rend; r += 2) {
    float v = preout[(size_t)r * COUT + c];
    s += v; ss += v * v;
  }
  sh[t] = s; sh[256 + t] = ss;
  __syncthreads();
  if (h == 0) {
    s  += sh[t + 128];
    ss += sh[256 + t + 128];
    part[blockIdx.x * 128 + c]         = s;
    part[32768 + blockIdx.x * 128 + c] = ss;
  }
}

__global__ void kstats_final(const float* __restrict__ part,
                             const float* __restrict__ gamma,
                             const float* __restrict__ beta,
                             float* __restrict__ ssbuf) {
  const int c = threadIdx.x;               // 128 threads
  float s = 0.f, q = 0.f;
  for (int p = 0; p < 256; ++p) {
    s += part[p * 128 + c];
    q += part[32768 + p * 128 + c];
  }
  const float inv = 1.f / (float)M;
  float mu  = s * inv;
  float var = q * inv - mu * mu;           // biased variance (ddof=0)
  float sc  = gamma[c] * rsqrtf(var + 1e-5f);
  ssbuf[c]       = sc;
  ssbuf[128 + c] = beta[c] - mu * sc;
}

// ---------------------------------------------------------------------------
// Kernel 5: BN affine + LeakyReLU(0.2), float4 vectorized.
// ---------------------------------------------------------------------------
__global__ __launch_bounds__(256) void kapply(const float* __restrict__ preout,
                                              const float* __restrict__ ssbuf,
                                              float* __restrict__ out) {
  int g = blockIdx.x * 256 + threadIdx.x;  // M*COUT/4 = 1048576 float4's
  int ci = g & 31;                          // channel quad index
  float4 v  = ((const float4*)preout)[g];
  float4 sc = ((const float4*)ssbuf)[ci];
  float4 sh = ((const float4*)(ssbuf + 128))[ci];
  float4 o;
  o.x = fmaf(v.x, sc.x, sh.x);
  o.y = fmaf(v.y, sc.y, sh.y);
  o.z = fmaf(v.z, sc.z, sh.z);
  o.w = fmaf(v.w, sc.w, sh.w);
  o.x = (o.x >= 0.f) ? o.x : 0.2f * o.x;
  o.y = (o.y >= 0.f) ? o.y : 0.2f * o.y;
  o.z = (o.z >= 0.f) ? o.z : 0.2f * o.z;
  o.w = (o.w >= 0.f) ? o.w : 0.2f * o.w;
  ((float4*)out)[g] = o;
}

// ---------------------------------------------------------------------------
extern "C" void kernel_launch(void* const* d_in, const int* in_sizes, int n_in,
                              void* d_out, int out_size, void* d_ws, size_t ws_size,
                              hipStream_t stream) {
  (void)in_sizes; (void)n_in; (void)out_size; (void)ws_size;
  const float* xyz    = (const float*)d_in[0];   // [8,4096,3]
  const float* feats  = (const float*)d_in[1];   // [8,4096,64]
  const float* kp     = (const float*)d_in[2];   // [15,3]
  const float* weight = (const float*)d_in[3];   // [15,64,128]
  const float* gamma  = (const float*)d_in[4];   // [128]
  const float* beta   = (const float*)d_in[5];   // [128]

  char* ws = (char*)d_ws;
  unsigned short* btile = (unsigned short*)(ws + WS_BTILE);
  int*   nidx   = (int*)(ws + WS_NIDX);
  float* preout = (float*)(ws + WS_PREOUT);
  float* part   = (float*)(ws + WS_PART);
  float* ssbuf  = (float*)(ws + WS_SS);
  float* outp   = (float*)d_out;

  krepack<<<512, 256, 0, stream>>>(weight, btile);
  kball<<<M / 8, 256, 0, stream>>>(xyz, nidx);
  kkpconv<<<M / 16, 256, LDS_BYTES, stream>>>(xyz, feats, kp, nidx, btile, preout);
  kstats_part<<<256, 256, 0, stream>>>(preout, part);
  kstats_final<<<1, 128, 0, stream>>>(part, gamma, beta, ssbuf);
  kapply<<<(M * COUT / 4) / 256, 256, 0, stream>>>(preout, ssbuf, outp);
}
